// GGNNWithTopoAlpha_8813272892055
// MI455X (gfx1250) — compile-verified, matrix operand order checked by reading, not run on hardware
//
#include <hip/hip_runtime.h>

// ---------------------------------------------------------------------------
// GGNNWithTopoAlpha for MI455X (gfx1250): bf16 WMMA GEMM pipeline with
// async global->LDS double-buffered B-operand staging.
//   B=4, N=2048, HD=256, E=4096, T=1024, 5 propagation steps.
// Topo residual refactored to avoid the ExE L1 build:
//   dH = B1 (B1^T(B1 Ef) + B2(B2^T Ef)),  Ef = B1^T h.
// All GEMMs: C(f32) = A(bf16,row-major) * B, with B given as B^T row-major.
// ---------------------------------------------------------------------------

#define BB  4
#define NN  2048
#define HD  256
#define EE  4096
#define TT  1024
#define STEPS 5

typedef __bf16 bf16_t;
typedef __attribute__((ext_vector_type(8)))  __bf16 bf16x8;
typedef __attribute__((ext_vector_type(16))) __bf16 bf16x16;
typedef __attribute__((ext_vector_type(8)))  float  v8f;

// ---------------------------------------------------------------------------
// TN bf16 WMMA GEMM: C[M,Nc] = A[M,K] * B[K,Nc]  (+C if beta)
//   A:  row-major bf16, leading dim lda, per-batch stride sA (0 = shared)
//   BT: B^T row-major bf16 [Nc,K], leading dim ldbt, stride sBT
//   C:  f32, leading dim ldc, stride sC
// Block = 128 threads = 4 waves stacked in M: block tile 256x64,
// wave tile 64x64, K-step 32.  The 64x32 B^T slab is shared by all 4 waves
// and staged through LDS with global_load_async_to_lds_b128 (ASYNCcnt),
// double-buffered.  A fragments stream directly from global (b128 loads).
// All M %256 == 0, Nc %64 == 0, K %32 == 0 in this pipeline: no bounds checks.
// ---------------------------------------------------------------------------
__global__ __launch_bounds__(128)
void gemm_bf16_tn(const bf16_t* __restrict__ A, long sA, int lda,
                  const bf16_t* __restrict__ BT, long sBT, int ldbt,
                  float* __restrict__ C, long sC, int ldc,
                  int M, int Nc, int K, int beta)
{
    __shared__ bf16_t bsh[2][64 * 32];   // two 4KB B^T slabs: 64 rows x 32 K

    A  += (long)blockIdx.z * sA;
    BT += (long)blockIdx.z * sBT;
    C  += (long)blockIdx.z * sC;

    const int lane = threadIdx.x & 31;
    const int wave = threadIdx.x >> 5;           // 0..3
    const int row0 = blockIdx.y * 256 + wave * 64;
    const int col0 = blockIdx.x * 64;
    const int lr   = lane & 15;                  // M (A) / N (B) in fragment
    const int half = lane >> 4;                  // K half-group select

    // async-copy mapping: wave w copies B^T tile rows [w*16, w*16+16);
    // one issue moves 8 rows (32 lanes x 16B), so 2 issues per wave/K-step.
    const int crow  = wave * 16 + (lane >> 2);   // 0..63 (rows 0..7 of group)
    const int cbyte = (lane & 3) * 16;           // byte offset within 64B row
    const unsigned ldsrow = (unsigned)(crow * 64 + cbyte);
    const unsigned lbuf0 = ((unsigned)(uintptr_t)(&bsh[0][0])) + ldsrow;
    const unsigned lbuf1 = ((unsigned)(uintptr_t)(&bsh[1][0])) + ldsrow;

    v8f acc[4][4];
#pragma unroll
    for (int i = 0; i < 4; ++i)
#pragma unroll
        for (int j = 0; j < 4; ++j)
            acc[i][j] = (v8f){0.f,0.f,0.f,0.f,0.f,0.f,0.f,0.f};

    // issue async copy of the 64x32 B^T slab at K offset kk into buffer `buf`
    auto issue_b = [&](int buf, int kk) {
        const char* g0 = (const char*)(BT + (long)(col0 + crow) * ldbt + kk)
                         + cbyte;
        const char* g1 = g0 + (long)8 * ldbt * 2;      // +8 rows
        unsigned l0 = buf ? lbuf1 : lbuf0;
        unsigned l1 = l0 + 8 * 64;
        asm volatile("global_load_async_to_lds_b128 %0, %1, off"
                     :: "v"(l0), "v"((unsigned long long)(uintptr_t)g0)
                     : "memory");
        asm volatile("global_load_async_to_lds_b128 %0, %1, off"
                     :: "v"(l1), "v"((unsigned long long)(uintptr_t)g1)
                     : "memory");
    };

    issue_b(0, 0);
    int cur = 0;
    for (int kk = 0; kk < K; kk += 32) {
        // own async writes done -> barrier -> everyone's writes visible
        asm volatile("s_wait_asynccnt 0x0" ::: "memory");
        __syncthreads();
        if (kk + 32 < K) {
            issue_b(cur ^ 1, kk + 32);                 // prefetch next slab
            __builtin_prefetch(A + (long)(row0 + lr) * lda + kk + 32, 0, 1);
        }
        const bf16_t* bbase = &bsh[cur][0];
        bf16x16 af[4], bfr[4];
#pragma unroll
        for (int i = 0; i < 4; ++i) {
            // A fragment (16-bit 16x32 layout): two contiguous 16B loads
            const bf16_t* pa = A + (long)(row0 + i * 16 + lr) * lda
                               + kk + half * 8;
            bf16x8 alo = *(const bf16x8*)(pa);
            bf16x8 ahi = *(const bf16x8*)(pa + 16);
            af[i] = __builtin_shufflevector(alo, ahi,
                      0,1,2,3,4,5,6,7,8,9,10,11,12,13,14,15);
            // B fragment from LDS slab (ds_load_b128 x2)
            const bf16_t* pb = bbase + (i * 16 + lr) * 32 + half * 8;
            bf16x8 blo = *(const bf16x8*)(pb);
            bf16x8 bhi = *(const bf16x8*)(pb + 16);
            bfr[i] = __builtin_shufflevector(blo, bhi,
                      0,1,2,3,4,5,6,7,8,9,10,11,12,13,14,15);
        }
#pragma unroll
        for (int i = 0; i < 4; ++i)
#pragma unroll
            for (int j = 0; j < 4; ++j)
                acc[i][j] = __builtin_amdgcn_wmma_f32_16x16x32_bf16(
                    false, af[i], false, bfr[j], (short)0, acc[i][j],
                    false, false);
        cur ^= 1;
    }

    // C/D layout: VGPR r -> m = r + 8*half, n = lane&15
#pragma unroll
    for (int i = 0; i < 4; ++i)
#pragma unroll
        for (int j = 0; j < 4; ++j)
#pragma unroll
            for (int r = 0; r < 8; ++r) {
                long idx = (long)(row0 + i * 16 + r + half * 8) * ldc
                           + col0 + j * 16 + lr;
                float v = acc[i][j][r];
                C[idx] = beta ? (C[idx] + v) : v;
            }
}

// ---------------------------------------------------------------------------
// f32 [R,C] -> bf16 [C,R] transpose+convert (R,C multiples of 32), batched z.
// ---------------------------------------------------------------------------
__global__ __launch_bounds__(256)
void transpose_cvt_kernel(const float* __restrict__ in, bf16_t* __restrict__ out,
                          int R, int C)
{
    __shared__ float tile[32][33];
    const long bo = (long)blockIdx.z * (long)R * C;
    const int c0 = blockIdx.x * 32, r0 = blockIdx.y * 32;
    const int tx = threadIdx.x, ty = threadIdx.y;  // block (32,8)
    for (int j = ty; j < 32; j += 8)
        tile[j][tx] = in[bo + (long)(r0 + j) * C + c0 + tx];
    __syncthreads();
    for (int j = ty; j < 32; j += 8)
        out[bo + (long)(c0 + j) * R + r0 + tx] = (bf16_t)tile[tx][j];
}

__global__ void cvt_f32_bf16_kernel(const float* __restrict__ in,
                                    bf16_t* __restrict__ out, long n)
{
    long i = (long)blockIdx.x * blockDim.x + threadIdx.x;
    long stride = (long)gridDim.x * blockDim.x;
    for (; i < n; i += stride) out[i] = (bf16_t)in[i];
}

// h0 = relu(attr * W_i + b_i)
__global__ void init_h_kernel(const float* __restrict__ attr,
                              const float* __restrict__ W_i,
                              const float* __restrict__ b_i,
                              float* __restrict__ h, long total)
{
    long i = (long)blockIdx.x * blockDim.x + threadIdx.x;
    long stride = (long)gridDim.x * blockDim.x;
    for (; i < total; i += stride) {
        int  hd = (int)(i & (HD - 1));
        long bn = i >> 8;
        float v = attr[bn] * W_i[hd] + b_i[hd];
        h[i] = v > 0.f ? v : 0.f;
    }
}

// GRU gates (torch order r,z,n) with fused biases; in-place h update.
__global__ void gru_kernel(const float* __restrict__ gx,
                           const float* __restrict__ gh,
                           const float* __restrict__ b_ih,
                           const float* __restrict__ b_hh,
                           float* __restrict__ h, long total)
{
    long i = (long)blockIdx.x * blockDim.x + threadIdx.x;
    long stride = (long)gridDim.x * blockDim.x;
    for (; i < total; i += stride) {
        int  hd = (int)(i & (HD - 1));
        long bn = i >> 8;
        const float* px = gx + bn * (3 * HD);
        const float* ph = gh + bn * (3 * HD);
        float xr = px[hd]          + b_ih[hd];
        float xz = px[HD + hd]     + b_ih[HD + hd];
        float xn = px[2 * HD + hd] + b_ih[2 * HD + hd];
        float hr = ph[hd]          + b_hh[hd];
        float hz = ph[HD + hd]     + b_hh[HD + hd];
        float hn = ph[2 * HD + hd] + b_hh[2 * HD + hd];
        float r = 1.f / (1.f + __expf(-(xr + hr)));
        float z = 1.f / (1.f + __expf(-(xz + hz)));
        float n = tanhf(xn + r * hn);
        h[i] = (1.f - z) * n + z * h[i];
    }
}

// h += alpha * LayerNorm(relu(proj)); one wave32 per row of 256.
__global__ __launch_bounds__(256)
void ln_alpha_kernel(const float* __restrict__ proj,
                     const float* __restrict__ gamma,
                     const float* __restrict__ beta,
                     const float* __restrict__ alpha_raw,
                     float* __restrict__ h, int rows)
{
    const int wave = threadIdx.x >> 5, lane = threadIdx.x & 31;
    const int row = blockIdx.x * 8 + wave;
    if (row >= rows) return;
    const float* pr = proj + (long)row * HD;
    float v[8], s = 0.f, s2 = 0.f;
#pragma unroll
    for (int i = 0; i < 8; ++i) {
        float x = pr[lane * 8 + i];
        x = x > 0.f ? x : 0.f;
        v[i] = x; s += x; s2 += x * x;
    }
    for (int m = 16; m; m >>= 1) {
        s  += __shfl_xor(s,  m, 32);
        s2 += __shfl_xor(s2, m, 32);
    }
    const float mu  = s * (1.f / HD);
    const float var = s2 * (1.f / HD) - mu * mu;
    const float rs  = rsqrtf(var + 1e-5f);
    const float alpha = 0.2f / (1.f + __expf(-alpha_raw[0]));
    float* ph = h + (long)row * HD;
#pragma unroll
    for (int i = 0; i < 8; ++i) {
        int c = lane * 8 + i;
        ph[c] += alpha * ((v[i] - mu) * rs * gamma[c] + beta[c]);
    }
}

// anomaly[row] = dot(h[row,:], W_o) + b_o ; one wave32 per row.
__global__ __launch_bounds__(256)
void out_kernel(const float* __restrict__ h, const float* __restrict__ W_o,
                const float* __restrict__ b_o, float* __restrict__ out, int rows)
{
    const int wave = threadIdx.x >> 5, lane = threadIdx.x & 31;
    const int row = blockIdx.x * 8 + wave;
    if (row >= rows) return;
    const float* ph = h + (long)row * HD;
    float s = 0.f;
#pragma unroll
    for (int i = 0; i < 8; ++i)
        s += ph[lane * 8 + i] * W_o[lane * 8 + i];
    for (int m = 16; m; m >>= 1) s += __shfl_xor(s, m, 32);
    if (lane == 0) out[row] = s + b_o[0];
}

// ---------------------------------------------------------------------------
extern "C" void kernel_launch(void* const* d_in, const int* in_sizes, int n_in,
                              void* d_out, int out_size, void* d_ws, size_t ws_size,
                              hipStream_t stream)
{
    const float* attr    = (const float*)d_in[0];
    const float* adj     = (const float*)d_in[1];
    const float* W_i     = (const float*)d_in[2];
    const float* b_i     = (const float*)d_in[3];
    const float* W_ih    = (const float*)d_in[4];
    const float* b_ih    = (const float*)d_in[5];
    const float* W_hh    = (const float*)d_in[6];
    const float* b_hh    = (const float*)d_in[7];
    const float* W_o     = (const float*)d_in[8];
    const float* b_o     = (const float*)d_in[9];
    const float* B1      = (const float*)d_in[10];
    const float* B2      = (const float*)d_in[11];
    const float* proj_W  = (const float*)d_in[12];
    const float* ln_g    = (const float*)d_in[13];
    const float* ln_b    = (const float*)d_in[14];
    const float* a_raw   = (const float*)d_in[15];
    float* out = (float*)d_out;

    // ---- workspace carve-out (256B aligned slabs) ----
    char* p = (char*)d_ws;
    auto alloc = [&](size_t bytes) -> void* {
        void* r = (void*)p;
        p += (bytes + 255) & ~(size_t)255;
        return r;
    };
    float*  h       = (float*) alloc((size_t)BB*NN*HD*4);
    bf16_t* hT_bf   = (bf16_t*)alloc((size_t)BB*HD*NN*2);
    bf16_t* h_bf    = (bf16_t*)alloc((size_t)BB*NN*HD*2);
    float*  x       = (float*) alloc((size_t)BB*NN*2*HD*4);
    bf16_t* x_bf    = (bf16_t*)alloc((size_t)BB*NN*2*HD*2);
    float*  gx      = (float*) alloc((size_t)BB*NN*3*HD*4);
    float*  gh      = (float*) alloc((size_t)BB*NN*3*HD*4);
    bf16_t* A_bf    = (bf16_t*)alloc((size_t)BB*NN*NN*2);
    bf16_t* AT_bf   = (bf16_t*)alloc((size_t)BB*NN*NN*2);
    bf16_t* Wih_bf  = (bf16_t*)alloc((size_t)3*HD*2*HD*2);
    bf16_t* Whh_bf  = (bf16_t*)alloc((size_t)3*HD*HD*2);
    bf16_t* B1_bf   = (bf16_t*)alloc((size_t)NN*EE*2);
    bf16_t* B1T_bf  = (bf16_t*)alloc((size_t)EE*NN*2);
    bf16_t* B2_bf   = (bf16_t*)alloc((size_t)EE*TT*2);
    bf16_t* B2T_bf  = (bf16_t*)alloc((size_t)TT*EE*2);
    bf16_t* pW_bf   = (bf16_t*)alloc((size_t)HD*HD*2);
    float*  Ef      = (float*) alloc((size_t)BB*EE*HD*4);
    bf16_t* EfT_bf  = (bf16_t*)alloc((size_t)BB*HD*EE*2);
    float*  P       = (float*) alloc((size_t)BB*NN*HD*4);
    bf16_t* PT_bf   = (bf16_t*)alloc((size_t)BB*HD*NN*2);
    float*  Q       = (float*) alloc((size_t)BB*TT*HD*4);
    bf16_t* QT_bf   = (bf16_t*)alloc((size_t)BB*HD*TT*2);
    float*  Et      = (float*) alloc((size_t)BB*EE*HD*4);
    bf16_t* EtT_bf  = (bf16_t*)alloc((size_t)BB*HD*EE*2);
    float*  dH      = (float*) alloc((size_t)BB*NN*HD*4);
    bf16_t* dH_bf   = (bf16_t*)alloc((size_t)BB*NN*HD*2);
    float*  prj     = (float*) alloc((size_t)BB*NN*HD*4);

    auto cvt = [&](const float* src, bf16_t* dst, long n) {
        int blocks = (int)((n + 255) / 256); if (blocks > 8192) blocks = 8192;
        cvt_f32_bf16_kernel<<<blocks, 256, 0, stream>>>(src, dst, n);
    };
    auto trans = [&](const float* src, bf16_t* dst, int R, int C, int batch) {
        dim3 g(C / 32, R / 32, batch), b(32, 8);
        transpose_cvt_kernel<<<g, b, 0, stream>>>(src, dst, R, C);
    };
    auto gemm = [&](const bf16_t* A_, long sA, int lda,
                    const bf16_t* BT_, long sBT, int ldbt,
                    float* C_, long sC, int ldc,
                    int M, int Nc, int K, int beta, int batch) {
        dim3 g(Nc / 64, M / 256, batch), b(128);
        gemm_bf16_tn<<<g, b, 0, stream>>>(A_, sA, lda, BT_, sBT, ldbt,
                                          C_, sC, ldc, M, Nc, K, beta);
    };

    // ---- static operand conversion (once per call) ----
    cvt(adj,    A_bf,  (long)BB * NN * NN);
    trans(adj,  AT_bf, NN, NN, BB);
    cvt(W_ih,   Wih_bf, (long)3 * HD * 2 * HD);
    cvt(W_hh,   Whh_bf, (long)3 * HD * HD);
    cvt(B1,     B1_bf,  (long)NN * EE);
    trans(B1,   B1T_bf, NN, EE, 1);
    cvt(B2,     B2_bf,  (long)EE * TT);
    trans(B2,   B2T_bf, EE, TT, 1);
    cvt(proj_W, pW_bf,  (long)HD * HD);

    // ---- h0 = relu(linear_i(attr)) ----
    const long nBNH = (long)BB * NN * HD;
    init_h_kernel<<<8192, 256, 0, stream>>>(attr, W_i, b_i, h, nBNH);

    // ---- GGNN propagation ----
    for (int step = 0; step < STEPS; ++step) {
        cvt(h, h_bf, nBNH);
        trans(h, hT_bf, NN, HD, BB);
        // a_in  = A    @ h  -> x[:, :, 0:256]
        gemm(A_bf,  (long)NN*NN, NN, hT_bf, (long)HD*NN, NN,
             x,       (long)NN*2*HD, 2*HD, NN, HD, NN, 0, BB);
        // a_out = A^T @ h  -> x[:, :, 256:512]
        gemm(AT_bf, (long)NN*NN, NN, hT_bf, (long)HD*NN, NN,
             x + HD,  (long)NN*2*HD, 2*HD, NN, HD, NN, 0, BB);
        cvt(x, x_bf, (long)BB * NN * 2 * HD);
        // gx = x @ W_ih^T   (B^T row-major == W_ih row-major)
        gemm(x_bf,  (long)NN*2*HD, 2*HD, Wih_bf, 0, 2*HD,
             gx,      (long)NN*3*HD, 3*HD, NN, 3*HD, 2*HD, 0, BB);
        // gh = h @ W_hh^T
        gemm(h_bf,  (long)NN*HD, HD, Whh_bf, 0, HD,
             gh,      (long)NN*3*HD, 3*HD, NN, 3*HD, HD, 0, BB);
        gru_kernel<<<8192, 256, 0, stream>>>(gx, gh, b_ih, b_hh, h, nBNH);
    }

    // ---- topo residual: dH = B1 (B1^T(B1 Ef) + B2(B2^T Ef)), Ef = B1^T h ----
    cvt(h, h_bf, nBNH);
    trans(h, hT_bf, NN, HD, BB);
    gemm(B1T_bf, 0, NN, hT_bf, (long)HD*NN, NN,
         Ef, (long)EE*HD, HD, EE, HD, NN, 0, BB);            // E_feat [B,E,H]
    trans(Ef, EfT_bf, EE, HD, BB);
    gemm(B1_bf, 0, EE, EfT_bf, (long)HD*EE, EE,
         P, (long)NN*HD, HD, NN, HD, EE, 0, BB);             // P = B1 Ef
    gemm(B2T_bf, 0, EE, EfT_bf, (long)HD*EE, EE,
         Q, (long)TT*HD, HD, TT, HD, EE, 0, BB);             // Q = B2^T Ef
    trans(P, PT_bf, NN, HD, BB);
    trans(Q, QT_bf, TT, HD, BB);
    gemm(B1T_bf, 0, NN, PT_bf, (long)HD*NN, NN,
         Et, (long)EE*HD, HD, EE, HD, NN, 0, BB);            // Et  = B1^T P
    gemm(B2_bf, 0, TT, QT_bf, (long)HD*TT, TT,
         Et, (long)EE*HD, HD, EE, HD, TT, 1, BB);            // Et += B2 Q
    trans(Et, EtT_bf, EE, HD, BB);
    gemm(B1_bf, 0, EE, EtT_bf, (long)HD*EE, EE,
         dH, (long)NN*HD, HD, NN, HD, EE, 0, BB);            // dH = B1 Et
    cvt(dH, dH_bf, nBNH);
    gemm(dH_bf, (long)NN*HD, HD, pW_bf, 0, HD,
         prj, (long)NN*HD, HD, NN, HD, HD, 0, BB);           // proj

    // h += alpha * LN(relu(proj))
    ln_alpha_kernel<<<BB * NN / 8, 256, 0, stream>>>(prj, ln_g, ln_b, a_raw,
                                                     h, BB * NN);
    // anomaly = h @ W_o^T + b_o
    out_kernel<<<BB * NN / 8, 256, 0, stream>>>(h, W_o, b_o, out, BB * NN);

    (void)in_sizes; (void)n_in; (void)out_size; (void)ws_size;
}
